// Hypernetwork_79121887527510
// MI455X (gfx1250) — compile-verified, matrix operand order checked by reading, not run on hardware
//
#include <hip/hip_runtime.h>

// ---------------- problem constants (from reference) ----------------
static constexpr int T_IN    = 12;
static constexpr int HORIZON = 12;
static constexpr int H_FSIZE = 64;
static constexpr int HIDDEN  = 64;
static constexpr int N_NODE  = 1024;
static constexpr int BATCH   = 64;

static constexpr int KDIM = 36864;            // (CHEB_K*HIDDEN)^2
static constexpr int MDIM = N_NODE;           // 1024 rows of h
static constexpr int NDIM = HORIZON * H_FSIZE;// 768 cols of h
static constexpr int KSPLIT = 8;              // deterministic split-K
static constexpr int KSEG   = KDIM / KSPLIT;  // 4608 (144 steps of 32)

typedef __attribute__((ext_vector_type(16))) __bf16 v16bf;
typedef __attribute__((ext_vector_type(8)))  float  v8f;

// Load one WMMA bf16 fragment's per-lane data (16 values) from an f32 row.
// Per the CDNA5 16-bit A/B layouts, lane holds two contiguous 8-element K
// chunks: [k .. k+7] and [k+16 .. k+23]; caller folds the (lane>=16 ? +8 : 0)
// chunk select into `p`. Four global_load_b128 + packed cvt to bf16.
__device__ __forceinline__ v16bf load_frag_f32_as_bf16(const float* p) {
    const float4* q = (const float4*)p;
    float4 c0 = q[0];   // k+0 .. k+3
    float4 c1 = q[1];   // k+4 .. k+7
    float4 c2 = q[4];   // k+16 .. k+19
    float4 c3 = q[5];   // k+20 .. k+23
    v16bf r;
    r[0]  = (__bf16)c0.x; r[1]  = (__bf16)c0.y; r[2]  = (__bf16)c0.z; r[3]  = (__bf16)c0.w;
    r[4]  = (__bf16)c1.x; r[5]  = (__bf16)c1.y; r[6]  = (__bf16)c1.z; r[7]  = (__bf16)c1.w;
    r[8]  = (__bf16)c2.x; r[9]  = (__bf16)c2.y; r[10] = (__bf16)c2.z; r[11] = (__bf16)c2.w;
    r[12] = (__bf16)c3.x; r[13] = (__bf16)c3.y; r[14] = (__bf16)c3.z; r[15] = (__bf16)c3.w;
    return r;
}

#define WMMA_BF16(A, B, C) \
    __builtin_amdgcn_wmma_f32_16x16x32_bf16(false, (A), false, (B), (short)0, (C), false, false)

// ---------------------------------------------------------------------------
// Kernel 1: partial[z] = weights[:, zK:(z+1)K] @ W_hyp[:, zK:(z+1)K]^T
// grid = (NDIM/64, MDIM/64, KSPLIT), block = 128 (4 waves, 2x2 of 32x32 tiles)
// ---------------------------------------------------------------------------
__global__ __launch_bounds__(128)
void hyp_gemm_kernel(const float* __restrict__ weights,  // [MDIM, KDIM]
                     const float* __restrict__ W_hyp,    // [NDIM, KDIM]
                     float* __restrict__ partial)        // [KSPLIT, MDIM, NDIM]
{
    const int lane = threadIdx.x & 31;
    const int w    = threadIdx.x >> 5;

    const int m0 = blockIdx.y * 64 + (w >> 1) * 32;   // 32-row wave tile
    const int n0 = blockIdx.x * 64 + (w & 1) * 32;    // 32-col wave tile
    const int kbeg = blockIdx.z * KSEG;

    const int lr = lane & 15;            // row (A) / col (B) within 16
    const int kl = (lane >> 4) * 8;      // K-chunk select per ISA layout

    const float* A0 = weights + (size_t)(m0 + lr) * KDIM;
    const float* A1 = A0 + (size_t)16 * KDIM;
    const float* B0 = W_hyp  + (size_t)(n0 + lr) * KDIM;
    const float* B1 = B0 + (size_t)16 * KDIM;

    v8f c00 = {}, c01 = {}, c10 = {}, c11 = {};

    for (int kb = kbeg; kb < kbeg + KSEG; kb += 32) {
        const int k = kb + kl;
        // speculative prefetch a few k-steps ahead on the streamed operands
        __builtin_prefetch(A0 + k + 128, 0, 0);
        __builtin_prefetch(A1 + k + 128, 0, 0);

        v16bf a0 = load_frag_f32_as_bf16(A0 + k);
        v16bf a1 = load_frag_f32_as_bf16(A1 + k);
        v16bf b0 = load_frag_f32_as_bf16(B0 + k);
        v16bf b1 = load_frag_f32_as_bf16(B1 + k);

        c00 = WMMA_BF16(a0, b0, c00);
        c01 = WMMA_BF16(a0, b1, c01);
        c10 = WMMA_BF16(a1, b0, c10);
        c11 = WMMA_BF16(a1, b1, c11);
    }

    // Store: D layout — lane holds col n = lane%16, VGPR r holds row r (+8 for
    // lanes 16-31).
    float* P = partial + (size_t)blockIdx.z * MDIM * NDIM;
    const int col  = lr;
    const int rofs = (lane >> 4) * 8;
    const v8f* acc[2][2] = { { &c00, &c01 }, { &c10, &c11 } };
#pragma unroll
    for (int i = 0; i < 2; ++i) {
#pragma unroll
        for (int j = 0; j < 2; ++j) {
            float* dst = P + (size_t)(m0 + i * 16 + rofs) * NDIM + (n0 + j * 16 + col);
            const v8f& c = *acc[i][j];
#pragma unroll
            for (int r = 0; r < 8; ++r)
                dst[(size_t)r * NDIM] = c[r];
        }
    }
}

// ---------------------------------------------------------------------------
// Kernel 2: h = sum_z partial[z] + b_hyp  (deterministic split-K reduction)
// ---------------------------------------------------------------------------
__global__ __launch_bounds__(256)
void hyp_reduce_kernel(const float* __restrict__ partial,
                       const float* __restrict__ b_hyp,
                       float* __restrict__ h)
{
    const int idx = blockIdx.x * 256 + threadIdx.x;   // < MDIM*NDIM
    float s = b_hyp[idx % NDIM];
#pragma unroll
    for (int z = 0; z < KSPLIT; ++z)
        s += partial[(size_t)z * MDIM * NDIM + idx];
    h[idx] = s;
}

// ---------------------------------------------------------------------------
// Kernel 3: out[b,t,n] = sum_f x[b,n,f] * h[n,t,f],  x = output[:, T_IN-1]
// One node per block; 4 waves cover the 64 batch rows (16 each); two K=32
// WMMA steps; only cols t < HORIZON are stored.
// grid = N_NODE, block = 128
// ---------------------------------------------------------------------------
__global__ __launch_bounds__(128)
void out_proj_kernel(const float* __restrict__ output,  // [B, T_IN, N, HIDDEN]
                     const float* __restrict__ h,       // [N, HORIZON, H_FSIZE]
                     float* __restrict__ out)           // [B, HORIZON, N]
{
    const int n    = blockIdx.x;
    const int lane = threadIdx.x & 31;
    const int w    = threadIdx.x >> 5;

    const int lr = lane & 15;
    const int kl = (lane >> 4) * 8;
    const int b0 = w * 16;

    // A row = batch b: output[((b*T_IN + T_IN-1)*N_NODE + n)*HIDDEN + f]
    const float* A = output +
        ((size_t)((b0 + lr) * T_IN + (T_IN - 1)) * N_NODE + n) * HIDDEN;
    // B col = horizon t (clamped; extra cols never stored)
    const int t = lr < HORIZON ? lr : HORIZON - 1;
    const float* B = h + ((size_t)n * HORIZON + t) * H_FSIZE;

    v8f c = {};
#pragma unroll
    for (int kb = 0; kb < HIDDEN; kb += 32) {
        v16bf a = load_frag_f32_as_bf16(A + kb + kl);
        v16bf b = load_frag_f32_as_bf16(B + kb + kl);
        c = WMMA_BF16(a, b, c);
    }

    if (lr < HORIZON) {
        const int rofs = (lane >> 4) * 8;
#pragma unroll
        for (int r = 0; r < 8; ++r) {
            const int b = b0 + rofs + r;
            out[((size_t)b * HORIZON + lr) * N_NODE + n] = c[r];
        }
    }
}

// ---------------------------------------------------------------------------
extern "C" void kernel_launch(void* const* d_in, const int* in_sizes, int n_in,
                              void* d_out, int out_size, void* d_ws, size_t ws_size,
                              hipStream_t stream)
{
    const float* output  = (const float*)d_in[0]; // (64,12,1024,64)
    const float* weights = (const float*)d_in[1]; // (1024,36864)
    const float* W_hyp   = (const float*)d_in[2]; // (768,36864)
    const float* b_hyp   = (const float*)d_in[3]; // (768,)
    float*       out     = (float*)d_out;         // (64,12,1024,1)

    float* partial = (float*)d_ws;                                  // KSPLIT*MDIM*NDIM
    float* hbuf    = partial + (size_t)KSPLIT * MDIM * NDIM;        // MDIM*NDIM

    dim3 g1(NDIM / 64, MDIM / 64, KSPLIT);       // 12 x 16 x 8 = 1536 blocks
    hyp_gemm_kernel<<<g1, dim3(128), 0, stream>>>(weights, W_hyp, partial);

    hyp_reduce_kernel<<<(MDIM * NDIM) / 256, 256, 0, stream>>>(partial, b_hyp, hbuf);

    out_proj_kernel<<<N_NODE, 128, 0, stream>>>(output, hbuf, out);
}
